// Attention_2817498546286
// MI455X (gfx1250) — compile-verified, matrix operand order checked by reading, not run on hardware
//
#include <hip/hip_runtime.h>
#include <hip/hip_bf16.h>

// ---------------------------------------------------------------------------
// MHA forward for MI455X (gfx1250): bf16 WMMA GEMMs + flash attention.
// B=4, N=2048, C=1024, H=16, D=64.
// ---------------------------------------------------------------------------

typedef __bf16 bf16;
typedef __attribute__((ext_vector_type(8)))  bf16  bf16x8;
typedef __attribute__((ext_vector_type(16))) bf16  bf16x16;
typedef __attribute__((ext_vector_type(8)))  float floatx8;
typedef int v4i __attribute__((__vector_size__(16)));  // matches builtin param

#define BATCH 4
#define SEQ   2048
#define EMB   1024
#define HEADS 16
#define HDIM  64
#define QKV_C 3072
#define MTOT  (BATCH * SEQ)   // 8192

// Async global->LDS DMA (ASYNCcnt-tracked) when the toolchain exposes it.
#if defined(__gfx1250__) && \
    __has_builtin(__builtin_amdgcn_global_load_async_to_lds_b128) && \
    __has_builtin(__builtin_amdgcn_s_wait_asynccnt)
#define USE_ASYNC_LDS 1
#else
#define USE_ASYNC_LDS 0
#endif

// Copy 16 bytes global -> LDS. Async path: global_load_async_to_lds_b128
// (no VGPR data movement). Fallback: b128 load + ds_store_b128.
__device__ inline void async_copy_b128(const bf16* g, bf16* l) {
#if USE_ASYNC_LDS
    __builtin_amdgcn_global_load_async_to_lds_b128((v4i*)g, (v4i*)l,
                                                   /*offset=*/0, /*cpol=*/0);
#else
    *reinterpret_cast<bf16x8*>(l) = *reinterpret_cast<const bf16x8*>(g);
#endif
}

__device__ inline void async_wait0() {
#if USE_ASYNC_LDS
    __builtin_amdgcn_s_wait_asynccnt(0);  // s_wait_asynccnt 0
#endif
}

__device__ inline bf16x16 make_frag16(bf16x8 lo, bf16x8 hi) {
    return __builtin_shufflevector(lo, hi, 0, 1, 2, 3, 4, 5, 6, 7,
                                           8, 9, 10, 11, 12, 13, 14, 15);
}

// Load a 16x32 bf16 A/B-style fragment from a row-major tile (LDS or global).
// Per ISA 7.12.2: lane L holds row (L&15); elems 0..7 at K = 8*(L>>4)+0..7,
// elems 8..15 at K = 16 + 8*(L>>4)+0..7.  (16B-aligned b128 loads.)
__device__ inline bf16x16 load_frag(const bf16* base, int pitch, int k0, int lane) {
    const int row = lane & 15;
    const int g8  = (lane >> 4) << 3;
    const bf16* r = base + (size_t)row * pitch + k0;
    bf16x8 lo = *reinterpret_cast<const bf16x8*>(r + g8);
    bf16x8 hi = *reinterpret_cast<const bf16x8*>(r + 16 + g8);
    return make_frag16(lo, hi);
}

__device__ inline floatx8 wmma_bf16(bf16x16 a, bf16x16 b, floatx8 c) {
    return __builtin_amdgcn_wmma_f32_16x16x32_bf16(
        /*neg_a=*/false, a, /*neg_b=*/false, b,
        /*c_mod=*/(short)0, c, /*reuse_a=*/false, /*reuse_b=*/false);
}

// ---------------------------------------------------------------------------
// fp32 -> bf16 conversion (bandwidth-trivial)
// ---------------------------------------------------------------------------
__global__ void cvt_f32_bf16_kernel(const float* __restrict__ src,
                                    bf16* __restrict__ dst, int n) {
    int i = blockIdx.x * blockDim.x + threadIdx.x;
    const int stride = gridDim.x * blockDim.x;
    for (; i < n; i += stride) dst[i] = (bf16)src[i];
}

// ---------------------------------------------------------------------------
// C[M,N] = A[M,K] @ Bw[N,K]^T + bias[N]
// 128 threads (4 waves). Block tile 128x128, each wave 64x64 (4x4 frags).
// Double-buffered 32-wide K slabs in LDS (padded pitch 40): async DMA for
// slab s+1 overlaps the 16 wmma of slab s; s_wait_asynccnt + one barrier
// per slab.  ds:wmma = 1:1 (16 ds_load_b128 feed 16 v_wmma per slab/wave).
// ---------------------------------------------------------------------------
__global__ __launch_bounds__(128)
void gemm_bf16_kernel(const bf16* __restrict__ A, const bf16* __restrict__ Bw,
                      const float* __restrict__ bias,
                      bf16* __restrict__ Cb, float* __restrict__ Cf,
                      int M, int Nn, int K) {
    __shared__ bf16 As[2][128 * 40];
    __shared__ bf16 Bs[2][128 * 40];

    const int tid  = threadIdx.x;
    const int lane = tid & 31;
    const int wave = tid >> 5;
    const int m0 = blockIdx.y * 128;
    const int n0 = blockIdx.x * 128;
    const int wm = (wave & 1) * 64;   // wave's M offset in block tile
    const int wn = (wave >> 1) * 64;  // wave's N offset in block tile
    const int nl = lane & 15;
    const int g  = lane >> 4;

    // Staging geometry: 512 b128 chunks per matrix per slab, 4 per thread.
    // Chunk t: row r_t = (tid>>2) + t*32, col c = (tid&3)*8 (constant).
    const int rbase = tid >> 2;
    const int cofs  = (tid & 3) << 3;
    const bf16* aptr[4];
    const bf16* bptr[4];
    int         lofs[4];
#pragma unroll
    for (int t = 0; t < 4; ++t) {
        const int r = rbase + t * 32;
        aptr[t] = A  + (size_t)(m0 + r) * K + cofs;
        bptr[t] = Bw + (size_t)(n0 + r) * K + cofs;
        lofs[t] = r * 40 + cofs;
    }

    floatx8 acc[4][4];
#pragma unroll
    for (int j = 0; j < 4; ++j) {
        const float bv = bias[n0 + wn + j * 16 + nl];
#pragma unroll
        for (int i = 0; i < 4; ++i)
#pragma unroll
            for (int e = 0; e < 8; ++e) acc[i][j][e] = bv;  // bias folded in
    }

    // Prologue: slab 0 -> buffer 0 via async DMA.
#pragma unroll
    for (int t = 0; t < 4; ++t) {
        async_copy_b128(aptr[t], &As[0][lofs[t]]);  aptr[t] += 32;
        async_copy_b128(bptr[t], &Bs[0][lofs[t]]);  bptr[t] += 32;
    }
    async_wait0();
    __syncthreads();

    const int nslab = K >> 5;
    for (int s = 0; s < nslab; ++s) {
        const int buf = s & 1;
        // DMA slab s+1 into the other buffer while wmma consumes this one.
        if (s + 1 < nslab) {
#pragma unroll
            for (int t = 0; t < 4; ++t) {
                async_copy_b128(aptr[t], &As[buf ^ 1][lofs[t]]);
                async_copy_b128(bptr[t], &Bs[buf ^ 1][lofs[t]]);
                if (s + 2 < nslab) {  // prefetch_b8 for slab s+2
                    __builtin_prefetch(aptr[t] + 32, 0, 1);
                    __builtin_prefetch(bptr[t] + 32, 0, 1);
                }
                aptr[t] += 32;
                bptr[t] += 32;
            }
        }

        // 16 wmma from the current buffer.
        bf16x16 af[4], bfr[4];
#pragma unroll
        for (int i = 0; i < 4; ++i)
            af[i] = load_frag(&As[buf][(wm + i * 16) * 40], 40, 0, lane);
#pragma unroll
        for (int j = 0; j < 4; ++j)
            bfr[j] = load_frag(&Bs[buf][(wn + j * 16) * 40], 40, 0, lane);
#pragma unroll
        for (int i = 0; i < 4; ++i)
#pragma unroll
            for (int j = 0; j < 4; ++j)
                acc[i][j] = wmma_bf16(af[i], bfr[j], acc[i][j]);

        async_wait0();     // this thread's DMA for slab s+1 landed in LDS
        __syncthreads();   // publish to all waves
    }

#pragma unroll
    for (int i = 0; i < 4; ++i)
#pragma unroll
        for (int j = 0; j < 4; ++j) {
            const int gn = n0 + wn + j * 16 + nl;
#pragma unroll
            for (int e = 0; e < 8; ++e) {
                const int gm = m0 + wm + i * 16 + e + 8 * g;
                const float v = acc[i][j][e];
                if (Cf) Cf[(size_t)gm * Nn + gn] = v;
                else    Cb[(size_t)gm * Nn + gn] = (bf16)v;
            }
        }
}

// ---------------------------------------------------------------------------
// Flash attention: one block per (b, h, 64-query tile). 128 threads / 4 waves.
// Each wave owns a 16-row Q strip (2 A-frags in registers, K-dim = 64).
// Streams 64-key tiles: S = (Q K^T)*scale via WMMA, online softmax in LDS,
// O += P V via WMMA (V staged transposed in LDS; K staged via async DMA).
// ---------------------------------------------------------------------------
__global__ __launch_bounds__(128)
void flash_attn_kernel(const bf16* __restrict__ qkv, bf16* __restrict__ O) {
    __shared__ bf16  Ks[64 * 72];   // [key][d]
    __shared__ bf16  Vt[64 * 72];   // [d][key]  (transposed on store)
    __shared__ bf16  Ps[64 * 72];   // [q][key]  exp(S - m)
    __shared__ float Sf[64 * 68];   // [q][key]  raw scores
    __shared__ float mrow[64], lrow[64], srow[64];

    const int tid  = threadIdx.x;
    const int lane = tid & 31;
    const int wave = tid >> 5;
    const int bh = blockIdx.y;
    const int b  = bh >> 4;
    const int h  = bh & 15;
    const int q0 = blockIdx.x * 64;
    const int nl = lane & 15;
    const int g  = lane >> 4;
    const float scale = 0.125f;     // 1/sqrt(64)

    const size_t rs = QKV_C;
    const bf16* qbase = qkv + (size_t)(b * SEQ + q0 + wave * 16) * rs + h * HDIM;
    const bf16* kbase = qkv + (size_t)(b * SEQ) * rs + EMB + h * HDIM;
    const bf16* vbase = kbase + EMB;

    // Per-thread staging pointers: 512 b128 chunks per tile, 4 per thread.
    // Chunk t: key_t = (tid>>3) + t*16, d col = (tid&7)*8 (constant).
    const int keyb = tid >> 3;
    const int dcol = (tid & 7) << 3;
    const bf16* kptr[4];
    const bf16* vptr[4];
#pragma unroll
    for (int t = 0; t < 4; ++t) {
        kptr[t] = kbase + (size_t)(keyb + t * 16) * rs + dcol;
        vptr[t] = vbase + (size_t)(keyb + t * 16) * rs + dcol;
    }

    // Q strip fragments: K-dim = 64 -> two 16x32 fragments, kept in registers.
    bf16x16 qf0 = load_frag(qbase, QKV_C, 0,  lane);
    bf16x16 qf1 = load_frag(qbase, QKV_C, 32, lane);

    floatx8 of[4];
#pragma unroll
    for (int j = 0; j < 4; ++j)
#pragma unroll
        for (int e = 0; e < 8; ++e) of[j][e] = 0.0f;

    if (tid < 64) { mrow[tid] = -1e30f; lrow[tid] = 0.0f; }
    __syncthreads();

    for (int kt = 0; kt < SEQ; kt += 64) {
        // Stage K via async DMA (row-major) and V via VGPRs (transposed).
#pragma unroll
        for (int t = 0; t < 4; ++t) {
            const int key = keyb + t * 16;
            async_copy_b128(kptr[t], &Ks[key * 72 + dcol]);
            const bf16x8 vv = *reinterpret_cast<const bf16x8*>(vptr[t]);
#pragma unroll
            for (int e = 0; e < 8; ++e) Vt[(dcol + e) * 72 + key] = vv[e];
            kptr[t] += 64 * rs;
            vptr[t] += 64 * rs;
        }
        async_wait0();
        __syncthreads();

        // S strip = scale * Q K^T  (8 wmma per wave), spilled to LDS (f32).
#pragma unroll
        for (int nf = 0; nf < 4; ++nf) {
            floatx8 s;
#pragma unroll
            for (int e = 0; e < 8; ++e) s[e] = 0.0f;
            s = wmma_bf16(qf0, load_frag(&Ks[(nf * 16) * 72], 72, 0,  lane), s);
            s = wmma_bf16(qf1, load_frag(&Ks[(nf * 16) * 72], 72, 32, lane), s);
#pragma unroll
            for (int e = 0; e < 8; ++e)
                Sf[(wave * 16 + e + 8 * g) * 68 + nf * 16 + nl] = s[e] * scale;
        }
        __syncthreads();

        // Online softmax: one thread per query row.
        if (tid < 64) {
            const int r = tid;
            const float m_old = mrow[r];
            float mx = m_old;
            for (int c = 0; c < 64; ++c) mx = fmaxf(mx, Sf[r * 68 + c]);
            const float fac = __expf(m_old - mx);
            float l = lrow[r] * fac;
            for (int c = 0; c < 64; ++c) {
                const float p = __expf(Sf[r * 68 + c] - mx);
                l += p;
                Ps[r * 72 + c] = (bf16)p;
            }
            mrow[r] = mx; lrow[r] = l; srow[r] = fac;
        }
        __syncthreads();

        // Rescale running O, then O += P V  (8 wmma per wave).
#pragma unroll
        for (int j = 0; j < 4; ++j)
#pragma unroll
            for (int e = 0; e < 8; ++e)
                of[j][e] *= srow[wave * 16 + e + 8 * g];
#pragma unroll
        for (int j = 0; j < 4; ++j) {
            of[j] = wmma_bf16(load_frag(&Ps[(wave * 16) * 72], 72, 0,  lane),
                              load_frag(&Vt[(j * 16)   * 72], 72, 0,  lane), of[j]);
            of[j] = wmma_bf16(load_frag(&Ps[(wave * 16) * 72], 72, 32, lane),
                              load_frag(&Vt[(j * 16)   * 72], 72, 32, lane), of[j]);
        }
        __syncthreads();
    }

    // Normalize by row sum and store O (bf16, 'b n (h d)' layout).
#pragma unroll
    for (int j = 0; j < 4; ++j)
#pragma unroll
        for (int e = 0; e < 8; ++e) {
            const int m = wave * 16 + e + 8 * g;
            const float v = of[j][e] / lrow[m];
            O[(size_t)(b * SEQ + q0 + m) * EMB + h * HDIM + j * 16 + nl] = (bf16)v;
        }
}

// ---------------------------------------------------------------------------
// Host-side orchestration
// ---------------------------------------------------------------------------
extern "C" void kernel_launch(void* const* d_in, const int* in_sizes, int n_in,
                              void* d_out, int out_size, void* d_ws, size_t ws_size,
                              hipStream_t stream) {
    const float* x    = (const float*)d_in[0];
    const float* wqkv = (const float*)d_in[1];
    const float* bqkv = (const float*)d_in[2];
    const float* wout = (const float*)d_in[3];
    const float* bout = (const float*)d_in[4];
    float* out = (float*)d_out;

    char* ws = (char*)d_ws;
    const size_t MB = 1024u * 1024u;
    bf16* xb    = (bf16*)(ws);                 // 16 MB:  x            (8192x1024)
    bf16* wqkvb = (bf16*)(ws + 16 * MB);       //  6 MB:  W_qkv        (3072x1024)
    bf16* woutb = (bf16*)(ws + 22 * MB);       //  2 MB:  W_out        (1024x1024)
    bf16* qkvb  = (bf16*)(ws + 24 * MB);       // 48 MB:  qkv          (8192x3072)
    bf16* ob    = (bf16*)(ws + 72 * MB);       // 16 MB:  attn output  (8192x1024)

    // 1) fp32 -> bf16 conversions
    cvt_f32_bf16_kernel<<<2048, 256, 0, stream>>>(x,    xb,    MTOT * EMB);
    cvt_f32_bf16_kernel<<<1024, 256, 0, stream>>>(wqkv, wqkvb, QKV_C * EMB);
    cvt_f32_bf16_kernel<<<512,  256, 0, stream>>>(wout, woutb, EMB * EMB);

    // 2) qkv = x @ W_qkv^T + b_qkv   (bf16 out)
    gemm_bf16_kernel<<<dim3(QKV_C / 128, MTOT / 128), 128, 0, stream>>>(
        xb, wqkvb, bqkv, qkvb, (float*)nullptr, MTOT, QKV_C, EMB);

    // 3) flash attention per (b, h, 64-query tile)
    flash_attn_kernel<<<dim3(SEQ / 64, BATCH * HEADS), 128, 0, stream>>>(qkvb, ob);

    // 4) out = O @ W_out^T + b_out   (fp32 out)
    gemm_bf16_kernel<<<dim3(EMB / 128, MTOT / 128), 128, 0, stream>>>(
        ob, woutb, bout, (bf16*)nullptr, out, MTOT, EMB, EMB);
}
